// LSTMDropout_19748259627794
// MI455X (gfx1250) — matrix operand order repaired and checked by reading, not compiled, not run on hardware
//
#include <hip/hip_runtime.h>
#include <hip/hip_bf16.h>

typedef __attribute__((ext_vector_type(16))) __bf16 v16bf;
typedef __attribute__((ext_vector_type(8)))  __bf16 v8bf;
typedef __attribute__((ext_vector_type(8)))  float  v8f;

static constexpr int kT = 512;   // timesteps
static constexpr int kB = 512;   // batch
static constexpr int kI = 128;   // input dim
static constexpr int kH = 256;   // hidden dim
static constexpr int kK = kI + kH;          // 384 total contraction
static constexpr int kChunks = kK / 32;     // 12 WMMA K-chunks
static constexpr int kJT = kH / 16;         // 16 hidden tiles
static constexpr float kScale = 1.0f / 0.9f;  // inverted dropout, p=0.1

// LDS pitches in bf16 elements, padded (+8 bf16 = 16B) to spread banks
static constexpr int kXPitch = kI + 8;   // 136
static constexpr int kHPitch = kH + 8;   // 264

// ---------------------------------------------------------------------------
// One-time (per launch) weight pre-swizzle: f32 (W: HxI, V: HxH) -> bf16
// fragment layout [gate][jt][chunk][lane][elem16], in this code's WMMA
// B-matrix (32x16 bf16) layout (contiguous k, same as A; not AMD's table order): lane = {half=K-group, n=lane&15},
// elem i -> K = chunk*32 + half*16 + i, column j = jt*16 + n.
// ---------------------------------------------------------------------------
__global__ void swizzle_weights(const float* __restrict__ Wf, const float* __restrict__ Vf,
                                const float* __restrict__ Wi, const float* __restrict__ Vi,
                                const float* __restrict__ Wc, const float* __restrict__ Vc,
                                const float* __restrict__ Wo, const float* __restrict__ Vo,
                                __bf16* __restrict__ wsW) {
    int idx = blockIdx.x * blockDim.x + threadIdx.x;
    const int total = 4 * kJT * kChunks * 512;     // 512 bf16 per fragment
    if (idx >= total) return;
    int e     = idx & 15;
    int lane  = (idx >> 4) & 31;
    int frag  = idx >> 9;
    int chunk = frag % kChunks;
    int jt    = (frag / kChunks) % kJT;
    int gate  = frag / (kChunks * kJT);
    int n     = lane & 15;
    int half  = lane >> 4;
    int k     = chunk * 32 + half * 16 + e;
    int j     = jt * 16 + n;
    const float* Ws[4] = {Wf, Wi, Wc, Wo};
    const float* Vs[4] = {Vf, Vi, Vc, Vo};
    float v = (k < kI) ? Ws[gate][j * kI + k] : Vs[gate][j * kH + (k - kI)];
    wsW[idx] = (__bf16)v;
}

// ---------------------------------------------------------------------------
// One LSTM timestep. Block = 128 threads = 4 waves; wave w computes gate w
// for the same 16(batch) x 16(hidden) tile, then the block does the
// elementwise c/h update locally. Grid = (B/16, H/16).
// ---------------------------------------------------------------------------
__global__ __launch_bounds__(128)
void lstm_step(const float* __restrict__ x_t,
               const float* __restrict__ h_prev,
               const float* c_prev,          // may alias c_next (in-place)
               float*       c_next,
               const __bf16* __restrict__ wsW,
               const float* __restrict__ b_f, const float* __restrict__ b_i,
               const float* __restrict__ b_c, const float* __restrict__ b_o,
               const float* __restrict__ m_f, const float* __restrict__ m_i,
               const float* __restrict__ m_c, const float* __restrict__ m_o,
               float* __restrict__ out_h) {
    __shared__ alignas(16) __bf16 lds_x[16 * kXPitch];
    __shared__ alignas(16) __bf16 lds_h[16 * kHPitch];
    __shared__ float lds_gate[4][256];

    const int tid  = threadIdx.x;
    const int lane = tid & 31;
    const int wave = tid >> 5;             // 0=f 1=i 2=c 3=o
    const int b0   = blockIdx.x * 16;
    const int j0   = blockIdx.y * 16;

    // ---- stage A operands to LDS as bf16 (coalesced f32 row reads) ----
    for (int i = tid; i < 16 * kI; i += 128) {
        int m = i >> 7, k = i & (kI - 1);
        lds_x[m * kXPitch + k] = (__bf16)x_t[(b0 + m) * kI + k];
    }
    for (int i = tid; i < 16 * kH; i += 128) {
        int m = i >> 8, k = i & (kH - 1);
        lds_h[m * kHPitch + k] = (__bf16)h_prev[(b0 + m) * kH + k];
    }
    __syncthreads();

    const float* bias = (wave == 0) ? b_f : (wave == 1) ? b_i : (wave == 2) ? b_c : b_o;
    const float* mask = (wave == 0) ? m_f : (wave == 1) ? m_i : (wave == 2) ? m_c : m_o;

    const int n    = lane & 15;
    const int half = lane >> 4;
    const int m    = lane & 15;            // A-matrix row owned by this lane

    // C/D layout: VGPR v holds (M = v + half*8, N = n) -> bias depends on N only
    float bj = bias[j0 + n];
    v8f acc;
#pragma unroll
    for (int v = 0; v < 8; ++v) acc[v] = bj;

    const __bf16* wbase =
        wsW + ((size_t)(wave * kJT + blockIdx.y) * kChunks) * 512 + lane * 16;

    union Frag { v16bf v; v8bf h[2]; };

#pragma unroll
    for (int c = 0; c < kChunks; ++c) {
        // A fragment: elems 0-15 <-> K = c*32 + half*16 + i
        const __bf16* arow = (c < 4) ? (lds_x + m * kXPitch + c * 32)
                                     : (lds_h + m * kHPitch + (c - 4) * 32);
        Frag a, b;
        a.h[0] = *(const v8bf*)(arow + half * 16);
        a.h[1] = *(const v8bf*)(arow + half * 16 + 8);
        const __bf16* bp = wbase + (size_t)c * 512;   // 32 lanes * 16 bf16 / frag
        b.h[0] = *(const v8bf*)(bp);
        b.h[1] = *(const v8bf*)(bp + 8);
        acc = __builtin_amdgcn_wmma_f32_16x16x32_bf16(
                  false, a.v, false, b.v, (short)0, acc, false, false);
    }

    // sigmoid + static dropout mask * 1/(1-p); share gate tile via LDS
#pragma unroll
    for (int v = 0; v < 8; ++v) {
        int mm = v + half * 8;
        float mk = mask[(size_t)(b0 + mm) * kH + (j0 + n)];
        float g  = (1.0f / (1.0f + __expf(-acc[v]))) * mk * kScale;
        lds_gate[wave][mm * 16 + n] = g;
    }
    __syncthreads();

    // elementwise LSTM update: 256 tile elements over 128 threads
#pragma unroll
    for (int e = tid; e < 256; e += 128) {
        int mm = e >> 4, nn = e & 15;
        size_t gi = (size_t)(b0 + mm) * kH + (j0 + nn);
        float fg = lds_gate[0][e];
        float ig = lds_gate[1][e];
        float gg = lds_gate[2][e];
        float og = lds_gate[3][e];
        float cp = c_prev[gi];
        float cn = fg * cp + ig * gg;
        c_next[gi] = cn;
        out_h[gi]  = og * tanhf(cn);
    }
}

__global__ void finalize_h(const float* __restrict__ last_h, float* __restrict__ h_out) {
    int i = blockIdx.x * blockDim.x + threadIdx.x;
    if (i < kB * kH) h_out[i] = last_h[i];
}

// ---------------------------------------------------------------------------
// Inputs (setup_inputs order):
//  0 x  1 h0  2 c0  3 W_f 4 V_f 5 W_i 6 V_i 7 W_c 8 V_c 9 W_o 10 V_o
//  11 b_f 12 b_i 13 b_c 14 b_o  15 mask_f 16 mask_i 17 mask_c 18 mask_o
// Output: outputs (T*B*H) ++ h (B*H) ++ c (B*H), f32.
// ---------------------------------------------------------------------------
extern "C" void kernel_launch(void* const* d_in, const int* in_sizes, int n_in,
                              void* d_out, int out_size, void* d_ws, size_t ws_size,
                              hipStream_t stream) {
    const float* x  = (const float*)d_in[0];
    const float* h0 = (const float*)d_in[1];
    const float* c0 = (const float*)d_in[2];
    const float* Wf = (const float*)d_in[3];
    const float* Vf = (const float*)d_in[4];
    const float* Wi = (const float*)d_in[5];
    const float* Vi = (const float*)d_in[6];
    const float* Wc = (const float*)d_in[7];
    const float* Vc = (const float*)d_in[8];
    const float* Wo = (const float*)d_in[9];
    const float* Vo = (const float*)d_in[10];
    const float* bf = (const float*)d_in[11];
    const float* bi = (const float*)d_in[12];
    const float* bc = (const float*)d_in[13];
    const float* bo = (const float*)d_in[14];
    const float* mf = (const float*)d_in[15];
    const float* mi = (const float*)d_in[16];
    const float* mc = (const float*)d_in[17];
    const float* mo = (const float*)d_in[18];

    float*  out = (float*)d_out;
    __bf16* wsW = (__bf16*)d_ws;                       // 4*16*12*512*2B = 768 KB

    const size_t BH = (size_t)kB * kH;
    float* c_state = out + (size_t)kT * BH + BH;       // final-c slot doubles as state

    // one-time per launch: pre-swizzle weights into WMMA B-fragment layout
    {
        int total = 4 * kJT * kChunks * 512;
        swizzle_weights<<<dim3((total + 255) / 256), 256, 0, stream>>>(
            Wf, Vf, Wi, Vi, Wc, Vc, Wo, Vo, wsW);
    }

    dim3 grid(kB / 16, kH / 16);   // 32 x 16 blocks
    for (int t = 0; t < kT; ++t) {
        const float* xt = x + (size_t)t * kB * kI;
        const float* hp = (t == 0) ? h0 : out + (size_t)(t - 1) * BH;
        const float* cp = (t == 0) ? c0 : c_state;
        float*       oh = out + (size_t)t * BH;
        lstm_step<<<grid, 128, 0, stream>>>(xt, hp, cp, c_state, wsW,
                                            bf, bi, bc, bo,
                                            mf, mi, mc, mo, oh);
    }

    // h_final = outputs[T-1]; c_final already lives in its slot (in-place state)
    finalize_h<<<dim3((int)((BH + 255) / 256)), 256, 0, stream>>>(
        out + (size_t)(kT - 1) * BH, out + (size_t)kT * BH);
}
